// FlashGQA_43688407335666
// MI455X (gfx1250) — matrix operand order repaired and checked by reading, not compiled, not run on hardware
//
#include <hip/hip_runtime.h>
#include <type_traits>

#define EMBED 2048
#define NHEADS 32
#define NKV 8
#define HD 64
#define BATCH 2
#define SEQ 2048
#define KVDIM (NKV * HD)      // 512
#define MROWS (BATCH * SEQ)   // 4096

typedef __attribute__((ext_vector_type(16))) __bf16 v16bf;
typedef __attribute__((ext_vector_type(8)))  float  v8f;

union AB32 { uint4 q[2]; v16bf v; };  // one 32-byte WMMA A/B fragment per lane

__device__ inline unsigned pack_bf2(float a, float b) {
  union { __bf16 h[2]; unsigned u; } x;
  x.h[0] = (__bf16)a; x.h[1] = (__bf16)b;
  return x.u;
}

// ---------------------------------------------------------------------------
// Generic Y[M,N] = X[M,K] @ W[N,K]^T.  X: float or bf16, W: float (converted
// to bf16 in LDS), Y: float or bf16.  Block = 256 threads (8 waves),
// tile = 256(M) x 64(N), BK = 32.  Each wave computes 32x64 (2 M-subtiles
// reusing each B fragment) -> 8 v_wmma per k-step against 12 ds_load_b128.
// ---------------------------------------------------------------------------
template <typename XT, typename OT>
__global__ __launch_bounds__(256) void gemm_xwt(const XT* __restrict__ X,
                                                const float* __restrict__ W,
                                                OT* __restrict__ Y,
                                                int M, int N, int K) {
  __shared__ alignas(16) __bf16 As[256][40];  // row stride 80B (16B multiple)
  __shared__ alignas(16) __bf16 Bs[64][40];

  const int t = threadIdx.x;
  const int wave = t >> 5, lane = t & 31;
  const int half = lane >> 4, l16 = lane & 15;
  const int m0 = blockIdx.y * 256;
  const int n0 = blockIdx.x * 64;

  v8f acc[2][4];
  for (int ms = 0; ms < 2; ++ms)
    for (int ns = 0; ns < 4; ++ns)
      for (int i = 0; i < 8; ++i) acc[ms][ns][i] = 0.0f;

  for (int kb = 0; kb < K; kb += 32) {
    // ---- stage X tile (256x32) as bf16: 1024 8-elem chunks, 4/thread ----
    for (int j = 0; j < 4; ++j) {
      int task = t + 256 * j;
      int row = task >> 2;
      int k0 = (task & 3) * 8;
      const XT* src = X + (size_t)(m0 + row) * K + kb + k0;
      if constexpr (std::is_same<XT, float>::value) {
        const float4* p = (const float4*)src;
        float4 a = p[0], b2 = p[1];
        *(uint4*)&As[row][k0] =
            make_uint4(pack_bf2(a.x, a.y), pack_bf2(a.z, a.w),
                       pack_bf2(b2.x, b2.y), pack_bf2(b2.z, b2.w));
      } else {
        *(uint4*)&As[row][k0] = *(const uint4*)src;  // already bf16
      }
    }
    // ---- stage W tile (64x32) as bf16: 256 chunks, 1/thread ----
    {
      int n = t >> 2;
      int k0 = (t & 3) * 8;
      const float4* p = (const float4*)(W + (size_t)(n0 + n) * K + kb + k0);
      float4 a = p[0], b2 = p[1];
      *(uint4*)&Bs[n][k0] =
          make_uint4(pack_bf2(a.x, a.y), pack_bf2(a.z, a.w),
                     pack_bf2(b2.x, b2.y), pack_bf2(b2.z, b2.w));
    }
    __syncthreads();

    // ---- fragments + WMMA (B reused across 2 M-subtiles) ----
    AB32 af[2];
    for (int ms = 0; ms < 2; ++ms) {
      const int arow = wave * 32 + ms * 16 + l16;
      af[ms].q[0] = *(const uint4*)&As[arow][half * 8];       // k = h*8..+7
      af[ms].q[1] = *(const uint4*)&As[arow][16 + half * 8];  // k = 16+h*8..+7
    }
    for (int ns = 0; ns < 4; ++ns) {
      AB32 bfrag;
      const int brow = ns * 16 + l16;
      bfrag.q[0] = *(const uint4*)&Bs[brow][half * 8];
      bfrag.q[1] = *(const uint4*)&Bs[brow][16 + half * 8];
      for (int ms = 0; ms < 2; ++ms)
        acc[ms][ns] = __builtin_amdgcn_wmma_f32_16x16x32_bf16(
            false, af[ms].v, false, bfrag.v, (short)0, acc[ms][ns], false,
            false);
    }
    __syncthreads();
  }

  for (int ms = 0; ms < 2; ++ms)
    for (int ns = 0; ns < 4; ++ns)
      for (int i = 0; i < 8; ++i) {
        int row = m0 + wave * 32 + ms * 16 + half * 8 + i;
        int col = n0 + ns * 16 + l16;
        if constexpr (std::is_same<OT, float>::value)
          Y[(size_t)row * N + col] = acc[ms][ns][i];
        else
          Y[(size_t)row * N + col] = (__bf16)acc[ms][ns][i];
      }
}

// ---------------------------------------------------------------------------
// RoPE in place on bf16 Q or K (layout (B*S, nheads*64)); one thread per
// (d, d+32) pair; fp32 math; rotate_half convention.
// ---------------------------------------------------------------------------
__global__ void rope_kernel(__bf16* __restrict__ T, const float* __restrict__ cosT,
                            const float* __restrict__ sinT, int nheads, int total) {
  int tid = blockIdx.x * blockDim.x + threadIdx.x;
  if (tid >= total) return;
  int d = tid & 31;
  int h = (tid >> 5) % nheads;
  int bs = tid / (32 * nheads);
  int s = bs % SEQ;
  size_t base = (size_t)bs * (nheads * HD) + h * HD + d;
  float c = cosT[s * HD + d];
  float sn = sinT[s * HD + d];
  float q1 = (float)T[base];
  float q2 = (float)T[base + 32];
  T[base]      = (__bf16)(q1 * c - q2 * sn);
  T[base + 32] = (__bf16)(q2 * c + q1 * sn);
}

// ---------------------------------------------------------------------------
// Flash attention with GQA + causal mask.  Block = 256 threads (8 waves),
// Q tile = 128 rows (16 per wave); 64-key K/V tiles staged in LDS and shared
// by all 8 waves (V transposed so P@V B-fragments are contiguous b128 LDS
// loads).  Online softmax in fp32; per-wave P relayout through LDS.
// ---------------------------------------------------------------------------
__global__ __launch_bounds__(256) void flash_attn(const __bf16* __restrict__ Qb,
                                                  const __bf16* __restrict__ Kb,
                                                  const __bf16* __restrict__ Vb,
                                                  __bf16* __restrict__ Ob) {
  __shared__ alignas(16) __bf16 Ks[64][72];      // [key][d], row 144B
  __shared__ alignas(16) __bf16 Vt[64][72];      // [d][key]
  __shared__ alignas(16) __bf16 Ps[8][16][64];   // per-wave P (D->A relayout)

  const int t = threadIdx.x;
  const int wave = t >> 5, lane = t & 31;
  const int half = lane >> 4, l16 = lane & 15;
  const int b = blockIdx.z, h = blockIdx.y;
  const int q0 = blockIdx.x * 128;
  const int kvh = h >> 2;  // repeat_interleave: q head h -> kv head h/4

  // Q fragments (A layout): rows q0 + wave*16 + l16, k = head dim
  AB32 qa[2];
  {
    const __bf16* qrow =
        Qb + (size_t)(b * SEQ + q0 + wave * 16 + l16) * EMBED + h * HD;
    int c0 = half * 8;
    qa[0].q[0] = *(const uint4*)(qrow + c0);
    qa[0].q[1] = *(const uint4*)(qrow + c0 + 16);
    qa[1].q[0] = *(const uint4*)(qrow + c0 + 32);
    qa[1].q[1] = *(const uint4*)(qrow + c0 + 48);
  }

  v8f oacc[4];
  float mrow[8], lrow[8];
  for (int s2 = 0; s2 < 4; ++s2)
    for (int i = 0; i < 8; ++i) oacc[s2][i] = 0.0f;
  for (int i = 0; i < 8; ++i) { mrow[i] = -3.0e38f; lrow[i] = 0.0f; }

  const int ntiles = 2 * blockIdx.x + 2;  // causal: keys <= q0+127
  for (int kt = 0; kt < ntiles; ++kt) {
    const int k0 = kt * 64;
    __syncthreads();  // previous-iteration readers done before restaging
    // ---- stage K tile and transposed V tile: 512 chunks, 2/thread ----
    for (int j = 0; j < 2; ++j) {
      int e = t + 256 * j;  // 0..511
      int key = e >> 3;
      int d0 = (e & 7) * 8;
      size_t grow = (size_t)(b * SEQ + k0 + key) * KVDIM + kvh * HD + d0;
      uint4 kv = *(const uint4*)(Kb + grow);
      *(uint4*)&Ks[key][d0] = kv;
      uint4 vv = *(const uint4*)(Vb + grow);
      const __bf16* vh = (const __bf16*)&vv;
      for (int q2 = 0; q2 < 8; ++q2) Vt[d0 + q2][key] = vh[q2];
    }
    __syncthreads();

    // ---- S = Q K^T (per wave: 16 x 64) ----
    v8f sc[4];
    for (int s2 = 0; s2 < 4; ++s2) {
      for (int i = 0; i < 8; ++i) sc[s2][i] = 0.0f;
      AB32 bf0, bf1;  // B(k=d, n=key) = Ks[key][d]
      const __bf16* krow = &Ks[s2 * 16 + l16][0];
      bf0.q[0] = *(const uint4*)(krow + half * 8);
      bf0.q[1] = *(const uint4*)(krow + half * 8 + 16);
      bf1.q[0] = *(const uint4*)(krow + 32 + half * 8);
      bf1.q[1] = *(const uint4*)(krow + 32 + half * 8 + 16);
      sc[s2] = __builtin_amdgcn_wmma_f32_16x16x32_bf16(
          false, qa[0].v, false, bf0.v, (short)0, sc[s2], false, false);
      sc[s2] = __builtin_amdgcn_wmma_f32_16x16x32_bf16(
          false, qa[1].v, false, bf1.v, (short)0, sc[s2], false, false);
    }

    // ---- scale + causal mask (only when tile can cross the diagonal) ----
    const bool diag = (k0 + 63) > (q0 + wave * 16);
    for (int s2 = 0; s2 < 4; ++s2)
      for (int i = 0; i < 8; ++i) {
        float v = sc[s2][i] * 0.125f;  // 1/sqrt(64)
        if (diag) {
          int qr = q0 + wave * 16 + half * 8 + i;
          int ky = k0 + s2 * 16 + l16;
          v = (ky <= qr) ? v : -1.0e30f;
        }
        sc[s2][i] = v;
      }

    // ---- online softmax (row stats live in the 16-lane half-group) ----
    float alpha[8];
    for (int i = 0; i < 8; ++i) {
      float v = fmaxf(fmaxf(sc[0][i], sc[1][i]), fmaxf(sc[2][i], sc[3][i]));
      for (int m2 = 8; m2 >= 1; m2 >>= 1) v = fmaxf(v, __shfl_xor(v, m2, 32));
      float mn = fmaxf(mrow[i], v);
      alpha[i] = __expf(mrow[i] - mn);
      mrow[i] = mn;
    }
    for (int i = 0; i < 8; ++i) {
      float rs = 0.0f;
      for (int s2 = 0; s2 < 4; ++s2) {
        float p = __expf(sc[s2][i] - mrow[i]);
        sc[s2][i] = p;
        rs += p;
      }
      for (int m2 = 8; m2 >= 1; m2 >>= 1) rs += __shfl_xor(rs, m2, 32);
      lrow[i] = lrow[i] * alpha[i] + rs;
      for (int s2 = 0; s2 < 4; ++s2) oacc[s2][i] *= alpha[i];
    }

    // ---- P: D-layout -> A-layout via wave-private LDS scratch ----
    for (int s2 = 0; s2 < 4; ++s2)
      for (int i = 0; i < 8; ++i)
        Ps[wave][half * 8 + i][s2 * 16 + l16] = (__bf16)sc[s2][i];
    __builtin_amdgcn_wave_barrier();
    asm volatile("s_wait_dscnt 0x0" ::: "memory");

    AB32 pa[2];
    {
      const __bf16* prow = &Ps[wave][l16][0];
      int c0 = half * 8;
      pa[0].q[0] = *(const uint4*)(prow + c0);
      pa[0].q[1] = *(const uint4*)(prow + c0 + 16);
      pa[1].q[0] = *(const uint4*)(prow + c0 + 32);
      pa[1].q[1] = *(const uint4*)(prow + c0 + 48);
    }

    // ---- O += P @ V : B(k=key, n=d) = Vt[d][key] ----
    for (int s2 = 0; s2 < 4; ++s2) {
      AB32 vf0, vf1;
      const __bf16* vrow = &Vt[s2 * 16 + l16][0];
      vf0.q[0] = *(const uint4*)(vrow + half * 8);
      vf0.q[1] = *(const uint4*)(vrow + half * 8 + 16);
      vf1.q[0] = *(const uint4*)(vrow + 32 + half * 8);
      vf1.q[1] = *(const uint4*)(vrow + 32 + half * 8 + 16);
      oacc[s2] = __builtin_amdgcn_wmma_f32_16x16x32_bf16(
          false, pa[0].v, false, vf0.v, (short)0, oacc[s2], false, false);
      oacc[s2] = __builtin_amdgcn_wmma_f32_16x16x32_bf16(
          false, pa[1].v, false, vf1.v, (short)0, oacc[s2], false, false);
    }
  }

  // ---- normalize + store O as (b, s, H*d) bf16 ----
  for (int s2 = 0; s2 < 4; ++s2)
    for (int i = 0; i < 8; ++i) {
      int row = b * SEQ + q0 + wave * 16 + half * 8 + i;
      int col = h * HD + s2 * 16 + l16;
      Ob[(size_t)row * EMBED + col] = (__bf16)(oacc[s2][i] / lrow[i]);
    }
}

// ---------------------------------------------------------------------------
extern "C" void kernel_launch(void* const* d_in, const int* in_sizes, int n_in,
                              void* d_out, int out_size, void* d_ws,
                              size_t ws_size, hipStream_t stream) {
  const float* x    = (const float*)d_in[0];
  const float* fcos = (const float*)d_in[1];
  const float* fsin = (const float*)d_in[2];
  const float* wq   = (const float*)d_in[3];
  const float* wk   = (const float*)d_in[4];
  const float* wv   = (const float*)d_in[5];
  const float* wo   = (const float*)d_in[6];
  float* out = (float*)d_out;

  char* ws = (char*)d_ws;
  const size_t qbytes = (size_t)MROWS * EMBED * 2;   // 16 MB
  const size_t kbytes = (size_t)MROWS * KVDIM * 2;   // 4 MB
  __bf16* Qb = (__bf16*)ws;
  __bf16* Kb = (__bf16*)(ws + qbytes);
  __bf16* Vb = (__bf16*)(ws + qbytes + kbytes);
  __bf16* Ob = (__bf16*)(ws + qbytes + 2 * kbytes);

  dim3 blk(256);
  gemm_xwt<float, __bf16><<<dim3(EMBED / 64, MROWS / 256), blk, 0, stream>>>(
      x, wq, Qb, MROWS, EMBED, EMBED);
  gemm_xwt<float, __bf16><<<dim3(KVDIM / 64, MROWS / 256), blk, 0, stream>>>(
      x, wk, Kb, MROWS, KVDIM, EMBED);
  gemm_xwt<float, __bf16><<<dim3(KVDIM / 64, MROWS / 256), blk, 0, stream>>>(
      x, wv, Vb, MROWS, KVDIM, EMBED);

  int totQ = MROWS * NHEADS * 32;
  rope_kernel<<<(totQ + 255) / 256, 256, 0, stream>>>(Qb, fcos, fsin, NHEADS, totQ);
  int totK = MROWS * NKV * 32;
  rope_kernel<<<(totK + 255) / 256, 256, 0, stream>>>(Kb, fcos, fsin, NKV, totK);

  flash_attn<<<dim3(SEQ / 128, NHEADS, BATCH), 256, 0, stream>>>(Qb, Kb, Vb, Ob);

  gemm_xwt<__bf16, float><<<dim3(EMBED / 64, MROWS / 256), blk, 0, stream>>>(
      Ob, wo, out, MROWS, EMBED, EMBED);
}
